// Attention_26482768347476
// MI455X (gfx1250) — matrix operand order repaired and checked by reading, not compiled, not run on hardware
//
#include <hip/hip_runtime.h>
#include <hip/hip_bf16.h>

// ---------------------------------------------------------------------------
// CDNA5 WMMA helpers (gfx1250, wave32)
// ---------------------------------------------------------------------------
typedef __attribute__((ext_vector_type(16))) __bf16 v16bf;
typedef __attribute__((ext_vector_type(8)))  float  v8f;
typedef int v4i_vec __attribute__((vector_size(16)));   // matches builtin proto

#define NUM_HEADS 12
#define DIM       768
#define HEAD_DIM  64
#define GRIDHW    64
#define HW        4096   // 64*64
#define N_QKV     2304   // 3*768
#define KP_TILES  (HW / 32)   // 128 key-pair tiles of 32 keys

// ---- gfx1250 async global->LDS path (ASYNCcnt), with portable fallback ----
#if defined(__has_builtin)
#if __has_builtin(__builtin_amdgcn_global_load_async_to_lds_b128) && \
    __has_builtin(__builtin_amdgcn_s_wait_asynccnt)
#define HAVE_ASYNC_LDS 1
#endif
#endif
#ifndef HAVE_ASYNC_LDS
#define HAVE_ASYNC_LDS 0
#endif

__device__ __forceinline__ void async_cp16(const void* gsrc, void* ldst) {
#if HAVE_ASYNC_LDS
    __builtin_amdgcn_global_load_async_to_lds_b128(
        (__attribute__((address_space(1))) v4i_vec*)gsrc,
        (__attribute__((address_space(3))) v4i_vec*)ldst,
        /*offset=*/0, /*cpol=*/0);
#else
    *(uint4*)ldst = *(const uint4*)gsrc;
#endif
}

template <int N>
__device__ __forceinline__ void wait_async() {
#if HAVE_ASYNC_LDS
    __builtin_amdgcn_s_wait_asynccnt(N);
#endif
}

__device__ __forceinline__ v8f wmma_bf16(v16bf a, v16bf b, v8f c) {
    // D = A(16x32) * B(32x16) + C(16x16), fp32 accumulate
    return __builtin_amdgcn_wmma_f32_16x16x32_bf16(
        /*neg_a=*/false, a, /*neg_b=*/false, b,
        /*c_mod=*/(short)0, c, /*reuse_a=*/false, /*reuse_b=*/false);
}

// A fragment (16x32 bf16, MxK): lane l -> row m=l%16; hi=l/16 selects K-halves.
// Elements 0..15 = cols [16*hi, 16*hi+16)
__device__ __forceinline__ v16bf load_a_frag(const __bf16* base, int ld) {
    int lane = threadIdx.x & 31;
    int m = lane & 15, hi = lane >> 4;
    const __bf16* row = base + (size_t)m * ld;
    v16bf a;
#pragma unroll
    for (int i = 0; i < 16; ++i) a[i] = row[16 * hi + i];
    return a;
}

// B fragment (32x16 bf16, KxN) loaded from a source where row n holds K
// contiguously: src[n*ld + k] == B[k][n].
// Lane l -> col n=l%16, K base 16*(l/16); 16 contiguous bf16.
__device__ __forceinline__ v16bf load_bT_frag(const __bf16* base, int ld) {
    int lane = threadIdx.x & 31;
    int n = lane & 15, kb = (lane >> 4) * 16;
    const __bf16* row = base + (size_t)n * ld + kb;
    v16bf b;
#pragma unroll
    for (int i = 0; i < 16; ++i) b[i] = row[i];
    return b;
}

// ---------------------------------------------------------------------------
// Precision conversion / transpose kernels
// ---------------------------------------------------------------------------
__global__ void f32_to_bf16_kernel(const float* __restrict__ in,
                                   __bf16* __restrict__ out, int n) {
    int i = blockIdx.x * blockDim.x + threadIdx.x;
    if (i < n) out[i] = (__bf16)in[i];
}

// in[rows][cols] (fp32) -> out[cols][rows] (bf16): makes weights N-major so B
// fragments are contiguous-per-lane loads.
__global__ void transpose_to_bf16_kernel(const float* __restrict__ in,
                                         __bf16* __restrict__ out,
                                         int rows, int cols) {
    int c = blockIdx.x * blockDim.x + threadIdx.x;
    int r = blockIdx.y;
    if (c < cols) out[(size_t)c * rows + r] = (__bf16)in[(size_t)r * cols + c];
}

// ---------------------------------------------------------------------------
// QKV GEMM: (4096 x 768) @ (768 x 2304) + bias, epilogue routes into per-head
// Q (row-major), K (row-major), V^T (d-major) bf16 buffers.
// Block = 128 threads (4 waves); block tile 64(M) x 64(N); wave tile 16 x 64.
// ---------------------------------------------------------------------------
__global__ __launch_bounds__(128) void qkv_gemm_kernel(
    const __bf16* __restrict__ xb,    // [4096][768]
    const __bf16* __restrict__ wt,    // [2304][768]  (N-major, K contiguous)
    const float*  __restrict__ bias,  // [2304]
    __bf16* __restrict__ qb,          // [12][4096][64]
    __bf16* __restrict__ kb,          // [12][4096][64]
    __bf16* __restrict__ vt)          // [12][64][4096]
{
    int wave = threadIdx.x >> 5;
    int lane = threadIdx.x & 31;
    int row0 = blockIdx.y * 64 + wave * 16;
    int col0 = blockIdx.x * 64;

    v8f acc[4] = {};
    for (int k0 = 0; k0 < DIM; k0 += 32) {
        v16bf a = load_a_frag(xb + (size_t)row0 * DIM + k0, DIM);
#pragma unroll
        for (int j = 0; j < 4; ++j) {
            v16bf b = load_bT_frag(wt + (size_t)(col0 + j * 16) * DIM + k0, DIM);
            acc[j] = wmma_bf16(a, b, acc[j]);
        }
    }

    int n_l = lane & 15, hi = lane >> 4;
#pragma unroll
    for (int j = 0; j < 4; ++j) {
        int col   = col0 + j * 16 + n_l;
        int which = col / DIM;          // 0=q 1=k 2=v
        int rem   = col % DIM;
        int head  = rem >> 6;
        int d     = rem & 63;
        float bv  = bias[col];
#pragma unroll
        for (int v = 0; v < 8; ++v) {
            int row = row0 + v + 8 * hi;
            __bf16 h = (__bf16)(acc[j][v] + bv);
            if (which == 0)
                qb[((size_t)head * HW + row) * HEAD_DIM + d] = h;
            else if (which == 1)
                kb[((size_t)head * HW + row) * HEAD_DIM + d] = h;
            else
                vt[((size_t)head * HEAD_DIM + d) * HW + row] = h;
        }
    }
}

// ---------------------------------------------------------------------------
// Decomposed rel-pos bias tables:
//   rel[head][qpos][k] = sum_d q[head][qpos][d] * rel_pos[(qc - k) + 63][d]
// axis==0 -> qc = qpos/64 (height), axis==1 -> qc = qpos%64 (width).
// ---------------------------------------------------------------------------
__global__ __launch_bounds__(64) void rel_bias_kernel(
    const __bf16* __restrict__ qb,       // [12][4096][64]
    const float*  __restrict__ rel_pos,  // [127][64]
    float* __restrict__ rel_out,         // [12][4096][64]
    int axis)
{
    int idx  = blockIdx.x;               // head*4096 + qpos
    int qpos = idx & (HW - 1);
    int qc   = (axis == 0) ? (qpos >> 6) : (qpos & 63);
    int k    = threadIdx.x;              // 0..63
    const __bf16* qrow = qb + (size_t)idx * HEAD_DIM;
    const float*  rp   = rel_pos + (size_t)(qc - k + GRIDHW - 1) * HEAD_DIM;
    float s = 0.f;
#pragma unroll
    for (int d = 0; d < HEAD_DIM; ++d) s += (float)qrow[d] * rp[d];
    rel_out[(size_t)idx * HEAD_DIM + k] = s;
}

// ---------------------------------------------------------------------------
// Flash attention: 4 waves per block, block covers (head, 64 q rows); waves
// share a K/V tile (32 keys) double-buffered in LDS via async global->LDS
// loads (ASYNCcnt). Each wave runs an independent online softmax for its own
// 16 q rows. Per 32-key tile per wave: 4 WMMAs for S, 4 WMMAs for PV.
// ---------------------------------------------------------------------------
__global__ __launch_bounds__(128) void flash_attn_kernel(
    const __bf16* __restrict__ qb,     // [12][4096][64]
    const __bf16* __restrict__ kb,     // [12][4096][64]
    const __bf16* __restrict__ vt,     // [12][64][4096]
    const float*  __restrict__ rel_h,  // [12][4096][64]
    const float*  __restrict__ rel_w,  // [12][4096][64]
    __bf16* __restrict__ out_hb)       // [4096][768] : [pos][head*64+d]
{
    __shared__ __bf16 s_k[2][32][HEAD_DIM];  // K rows (32 keys x 64 d)  8KB
    __shared__ __bf16 s_v[2][HEAD_DIM][32];  // V^T rows (64 d x 32 keys) 8KB
    __shared__ float  s_rh[64][64];          // 16KB
    __shared__ float  s_rw[64][64];          // 16KB
    __shared__ float  s_p[4][16][32];        // per-wave P relay, 8KB

    int head = blockIdx.y;
    int q0b  = blockIdx.x * 64;              // block q base (64 rows)
    int tid  = threadIdx.x;
    int wave = tid >> 5;
    int lane = tid & 31;
    int n_l  = lane & 15, hi = lane >> 4;
    int q0   = q0b + wave * 16;              // this wave's q base

    // Cache bias rows for the whole block's q tile.
    for (int i = tid; i < 64 * 64; i += 128) {
        int r = i >> 6, c = i & 63;
        size_t off = ((size_t)head * HW + (q0b + r)) * HEAD_DIM + c;
        s_rh[r][c] = rel_h[off];
        s_rw[r][c] = rel_w[off];
    }

    const __bf16* kt_base = kb + (size_t)head * HW * HEAD_DIM;   // [4096][64]
    const __bf16* vt_base = vt + (size_t)head * HEAD_DIM * HW;   // [64][4096]

    // Stage one 32-key tile (K: 4KB contiguous, V^T: 64 rows x 64B) into LDS.
    // 4 x 16B async copies per thread => ASYNCcnt +4 per wave per tile.
    auto stage = [&](int buf, int kp) {
        int kb0 = kp * 32;
        const char* ksrc = (const char*)(kt_base + (size_t)kb0 * HEAD_DIM);
        char* kdst = (char*)&s_k[buf][0][0];
        async_cp16(ksrc + tid * 16, kdst + tid * 16);
        async_cp16(ksrc + 2048 + tid * 16, kdst + 2048 + tid * 16);
#pragma unroll
        for (int c = tid; c < 256; c += 128) {       // 16B chunks of V tile
            int d = c >> 2, xo = (c & 3) * 16;
            const char* vsrc = (const char*)(vt_base + (size_t)d * HW + kb0) + xo;
            char* vdst = (char*)&s_v[buf][d][0] + xo;
            async_cp16(vsrc, vdst);
        }
    };

    const __bf16* qbase = qb + ((size_t)head * HW + q0) * HEAD_DIM;
    v16bf aq0 = load_a_frag(qbase + 0,  HEAD_DIM);
    v16bf aq1 = load_a_frag(qbase + 32, HEAD_DIM);

    v8f o[4] = {};
    float m_i[8], l_i[8];
#pragma unroll
    for (int v = 0; v < 8; ++v) { m_i[v] = -1e30f; l_i[v] = 0.f; }
    const float scale = 0.125f;  // 64^-0.5

    stage(0, 0);                                   // prologue prefetch
    for (int kp = 0; kp < KP_TILES; ++kp) {
        int cur = kp & 1;
        int kb0 = kp * 32;
        if (kp + 1 < KP_TILES) {                   // prefetch next tile
            stage(cur ^ 1, kp + 1);
            wait_async<4>();                       // current tile's 4 done
        } else {
            wait_async<0>();
        }
        __syncthreads();                           // tile visible to all waves

        // ---- S = Q K^T for the two 16-key halves (B frags from LDS) -------
        v8f s0 = {}, s1 = {};
        s0 = wmma_bf16(aq0, load_bT_frag(&s_k[cur][0][0],  HEAD_DIM), s0);
        s0 = wmma_bf16(aq1, load_bT_frag(&s_k[cur][0][32], HEAD_DIM), s0);
        s1 = wmma_bf16(aq0, load_bT_frag(&s_k[cur][16][0],  HEAD_DIM), s1);
        s1 = wmma_bf16(aq1, load_bT_frag(&s_k[cur][16][32], HEAD_DIM), s1);

        // ---- scale + decomposed rel-pos bias ------------------------------
        int j0 = kb0 + n_l, j1 = kb0 + 16 + n_l;
        int kh0 = j0 >> 6, kw0 = j0 & 63;
        int kh1 = j1 >> 6, kw1 = j1 & 63;
        float sv0[8], sv1[8];
#pragma unroll
        for (int v = 0; v < 8; ++v) {
            int r = wave * 16 + v + 8 * hi;        // block-local row
            sv0[v] = s0[v] * scale + s_rh[r][kh0] + s_rw[r][kw0];
            sv1[v] = s1[v] * scale + s_rh[r][kh1] + s_rw[r][kw1];
        }
        // ---- online softmax (row = 16 lanes within a half-wave) -----------
        float alpha[8];
#pragma unroll
        for (int v = 0; v < 8; ++v) {
            float x = fmaxf(sv0[v], sv1[v]);
#pragma unroll
            for (int off = 1; off < 16; off <<= 1)
                x = fmaxf(x, __shfl_xor(x, off, 32));
            float mn = fmaxf(m_i[v], x);
            alpha[v] = __expf(m_i[v] - mn);
            m_i[v]   = mn;
            sv0[v]   = __expf(sv0[v] - mn);
            sv1[v]   = __expf(sv1[v] - mn);
        }
#pragma unroll
        for (int v = 0; v < 8; ++v) {
            float x = sv0[v] + sv1[v];
#pragma unroll
            for (int off = 1; off < 16; off <<= 1)
                x += __shfl_xor(x, off, 32);
            l_i[v] = l_i[v] * alpha[v] + x;
        }
#pragma unroll
        for (int t = 0; t < 4; ++t)
#pragma unroll
            for (int v = 0; v < 8; ++v) o[t][v] *= alpha[v];

        // ---- re-lay P (D-layout) as an A fragment via per-wave LDS --------
#pragma unroll
        for (int v = 0; v < 8; ++v) {
            s_p[wave][v + 8 * hi][n_l]      = sv0[v];
            s_p[wave][v + 8 * hi][16 + n_l] = sv1[v];
        }
        __syncthreads();
        v16bf ap;
        {
            int m = n_l;
#pragma unroll
            for (int i = 0; i < 16; ++i) ap[i] = (__bf16)s_p[wave][m][16 * hi + i];
        }

        // ---- O += P V  (B frags from LDS V^T tile) ------------------------
#pragma unroll
        for (int t = 0; t < 4; ++t) {
            v16bf bv = load_bT_frag(&s_v[cur][t * 16][0], 32);
            o[t] = wmma_bf16(ap, bv, o[t]);
        }
        __syncthreads();   // all waves done reading this buffer before reuse
    }

    // ---- epilogue: normalize and scatter into [pos][head*64+d] ------------
#pragma unroll
    for (int t = 0; t < 4; ++t)
#pragma unroll
        for (int v = 0; v < 8; ++v) {
            int r = q0 + v + 8 * hi;
            int d = t * 16 + n_l;
            out_hb[(size_t)r * DIM + head * HEAD_DIM + d] =
                (__bf16)(o[t][v] / l_i[v]);
        }
}

// ---------------------------------------------------------------------------
// Projection GEMM: (4096 x 768) @ (768 x 768) + bias -> fp32 output
// ---------------------------------------------------------------------------
__global__ __launch_bounds__(128) void proj_gemm_kernel(
    const __bf16* __restrict__ xb,    // [4096][768]
    const __bf16* __restrict__ wt,    // [768][768] (N-major)
    const float*  __restrict__ bias,  // [768]
    float* __restrict__ out)          // [4096][768]
{
    int wave = threadIdx.x >> 5;
    int lane = threadIdx.x & 31;
    int row0 = blockIdx.y * 64 + wave * 16;
    int col0 = blockIdx.x * 64;

    v8f acc[4] = {};
    for (int k0 = 0; k0 < DIM; k0 += 32) {
        v16bf a = load_a_frag(xb + (size_t)row0 * DIM + k0, DIM);
#pragma unroll
        for (int j = 0; j < 4; ++j) {
            v16bf b = load_bT_frag(wt + (size_t)(col0 + j * 16) * DIM + k0, DIM);
            acc[j] = wmma_bf16(a, b, acc[j]);
        }
    }
    int n_l = lane & 15, hi = lane >> 4;
#pragma unroll
    for (int j = 0; j < 4; ++j) {
        int col  = col0 + j * 16 + n_l;
        float bv = bias[col];
#pragma unroll
        for (int v = 0; v < 8; ++v) {
            int row = row0 + v + 8 * hi;
            out[(size_t)row * DIM + col] = acc[j][v] + bv;
        }
    }
}

// ---------------------------------------------------------------------------
// Host launcher
// ---------------------------------------------------------------------------
extern "C" void kernel_launch(void* const* d_in, const int* in_sizes, int n_in,
                              void* d_out, int out_size, void* d_ws, size_t ws_size,
                              hipStream_t stream) {
    const float* x         = (const float*)d_in[0];  // [1][64][64][768]
    const float* qkv_w     = (const float*)d_in[1];  // [768][2304]
    const float* qkv_b     = (const float*)d_in[2];  // [2304]
    const float* proj_w    = (const float*)d_in[3];  // [768][768]
    const float* proj_b    = (const float*)d_in[4];  // [768]
    const float* rel_pos_h = (const float*)d_in[5];  // [127][64]
    const float* rel_pos_w = (const float*)d_in[6];  // [127][64]
    float* out = (float*)d_out;                      // [4096][768]

    // ---- workspace carving -------------------------------------------------
    char* ws = (char*)d_ws;
    size_t off = 0;
    auto alloc = [&](size_t bytes) {
        void* p = ws + off;
        off += (bytes + 255) & ~(size_t)255;
        return p;
    };
    __bf16* xb      = (__bf16*)alloc((size_t)HW * DIM * 2);          // x bf16
    __bf16* wqkv_t  = (__bf16*)alloc((size_t)N_QKV * DIM * 2);       // qkv_w^T bf16
    __bf16* wproj_t = (__bf16*)alloc((size_t)DIM * DIM * 2);         // proj_w^T bf16
    __bf16* qb      = (__bf16*)alloc((size_t)NUM_HEADS * HW * HEAD_DIM * 2);
    __bf16* kbuf    = (__bf16*)alloc((size_t)NUM_HEADS * HW * HEAD_DIM * 2);
    __bf16* vt      = (__bf16*)alloc((size_t)NUM_HEADS * HEAD_DIM * HW * 2);
    float*  relh    = (float*) alloc((size_t)NUM_HEADS * HW * HEAD_DIM * 4);
    float*  relw    = (float*) alloc((size_t)NUM_HEADS * HW * HEAD_DIM * 4);
    __bf16* out_hb  = (__bf16*)alloc((size_t)HW * DIM * 2);

    // 1) conversions
    {
        int n = HW * DIM;
        f32_to_bf16_kernel<<<(n + 255) / 256, 256, 0, stream>>>(x, xb, n);
    }
    transpose_to_bf16_kernel<<<dim3((N_QKV + 255) / 256, DIM), 256, 0, stream>>>(
        qkv_w, wqkv_t, DIM, N_QKV);
    transpose_to_bf16_kernel<<<dim3((DIM + 255) / 256, DIM), 256, 0, stream>>>(
        proj_w, wproj_t, DIM, DIM);

    // 2) fused QKV GEMM + head split
    qkv_gemm_kernel<<<dim3(N_QKV / 64, HW / 64), 128, 0, stream>>>(
        xb, wqkv_t, qkv_b, qb, kbuf, vt);

    // 3) decomposed rel-pos bias tables
    rel_bias_kernel<<<NUM_HEADS * HW, 64, 0, stream>>>(qb, rel_pos_h, relh, 0);
    rel_bias_kernel<<<NUM_HEADS * HW, 64, 0, stream>>>(qb, rel_pos_w, relw, 1);

    // 4) flash attention (4 waves share K/V tiles double-buffered in LDS)
    flash_attn_kernel<<<dim3(HW / 64, NUM_HEADS), 128, 0, stream>>>(
        qb, kbuf, vt, relh, relw, out_hb);

    // 5) output projection
    proj_gemm_kernel<<<dim3(DIM / 64, HW / 64), 128, 0, stream>>>(
        out_hb, wproj_t, proj_b, out);
}
